// Model_39676907887484
// MI455X (gfx1250) — matrix operand order repaired and checked by reading, not compiled, not run on hardware
//
#include <hip/hip_runtime.h>

#define B_  4
#define LQ_ 4096
#define LK_ 4096
#define H_  1024

typedef __bf16 bf16;
typedef __attribute__((ext_vector_type(16))) __bf16 v16bf;
typedef __attribute__((ext_vector_type(8)))  __bf16 v8bf;
typedef __attribute__((ext_vector_type(8)))  float  v8f;
typedef __attribute__((ext_vector_type(4)))  float  v4f;

// ---------- fragment loaders ----------
// A fragment (16x32 bf16 MxK): per lane two adjacent 16B chunks (contiguous 16 K).
__device__ __forceinline__ v16bf frag_a_bf16(const bf16* __restrict__ p) {
  v8bf lo = *(const v8bf*)p;
  v8bf hi = *(const v8bf*)(p + 8);
  return __builtin_shufflevector(lo, hi, 0, 1, 2, 3, 4, 5, 6, 7,
                                 8, 9, 10, 11, 12, 13, 14, 15);
}

// B fragment (32x16 bf16 KxN from row-per-N, contiguous-K memory):
// per lane one contiguous 32B chunk (32B aligned) -> single v16bf load.
__device__ __forceinline__ v16bf frag_b_bf16(const bf16* __restrict__ p) {
  return *(const v16bf*)p;
}

// A fragment from f32 source: two 32B chunks, native f32->bf16 casts
// (lowers to v_cvt_pk_bf16_f32).
__device__ __forceinline__ v16bf frag_a_f32(const float* __restrict__ p) {
  v4f x0 = *(const v4f*)(p + 0);
  v4f x1 = *(const v4f*)(p + 4);
  v4f x2 = *(const v4f*)(p + 8);
  v4f x3 = *(const v4f*)(p + 12);
  v16bf a;
#pragma unroll
  for (int i = 0; i < 4; ++i) {
    a[i]      = (bf16)x0[i];
    a[4 + i]  = (bf16)x1[i];
    a[8 + i]  = (bf16)x2[i];
    a[12 + i] = (bf16)x3[i];
  }
  return a;
}

// B fragment from f32 source: one contiguous 64B chunk.
__device__ __forceinline__ v16bf frag_b_f32(const float* __restrict__ p) {
  v4f x0 = *(const v4f*)(p + 0);
  v4f x1 = *(const v4f*)(p + 4);
  v4f x2 = *(const v4f*)(p + 8);
  v4f x3 = *(const v4f*)(p + 12);
  v16bf b;
#pragma unroll
  for (int i = 0; i < 4; ++i) {
    b[i]      = (bf16)x0[i];
    b[4 + i]  = (bf16)x1[i];
    b[8 + i]  = (bf16)x2[i];
    b[12 + i] = (bf16)x3[i];
  }
  return b;
}

#define WMMA_BF16(ACC, Afr, Bfr)                                            \
  (ACC) = __builtin_amdgcn_wmma_f32_16x16x32_bf16(false, (Afr), false,      \
                                                  (Bfr), (short)0, (ACC),   \
                                                  false, false)

// Per-wave register tile: 2 M-subtiles x 4 N-subtiles = 32x64 output.
#define MT 2
#define NT 4

// ============================================================
// Kernel 1: Y = X @ W^T + bias ; X:[16384,H] f32, W:[H,H] f32
// trans==0: Y row-major bf16 [16384,H]
// trans==1: Y per-batch transposed bf16 [B][H][LK]
// tiles: 512 (M) x 16 (N) = 8192 -> 1024 blocks x 8 waves.
// ============================================================
__global__ __launch_bounds__(256) void proj_kernel(
    const float* __restrict__ X, const float* __restrict__ W,
    const float* __restrict__ bias, bf16* __restrict__ Y, int trans) {
  const int lane = threadIdx.x & 31;
  const int wave = threadIdx.x >> 5;
  int tile = blockIdx.x * 8 + wave;   // [0, 8192)
  int tm = tile >> 4;                 // 512 M-tiles (32 rows)
  int tn = tile & 15;                 // 16  N-tiles (64 cols)
  int m0 = tm * 32, n0 = tn * 64;

  const int lrow = lane & 15;
  const int aoff = (lane & 16) ? 16 : 0;
  const int boff = (lane & 16) ? 16 : 0;
  const float* aBase[MT];
  const float* bBase[NT];
#pragma unroll
  for (int i = 0; i < MT; ++i)
    aBase[i] = X + (size_t)(m0 + 16 * i + lrow) * H_ + aoff;
#pragma unroll
  for (int j = 0; j < NT; ++j)
    bBase[j] = W + (size_t)(n0 + 16 * j + lrow) * H_ + boff;

  v8f acc[MT][NT] = {};
  for (int k = 0; k < H_; k += 32) {
    v16bf a[MT], bfr[NT];
#pragma unroll
    for (int i = 0; i < MT; ++i) a[i] = frag_a_f32(aBase[i] + k);
#pragma unroll
    for (int j = 0; j < NT; ++j) bfr[j] = frag_b_f32(bBase[j] + k);
#pragma unroll
    for (int i = 0; i < MT; ++i)
#pragma unroll
      for (int j = 0; j < NT; ++j) WMMA_BF16(acc[i][j], a[i], bfr[j]);
  }

  int n = lane & 15, hi = lane >> 4;
#pragma unroll
  for (int i = 0; i < MT; ++i) {
    int mbase = m0 + 16 * i + 8 * hi;
#pragma unroll
    for (int j = 0; j < NT; ++j) {
      int o = n0 + 16 * j + n;
      float bj = bias[o];
#pragma unroll
      for (int r = 0; r < 8; ++r) {
        float val = acc[i][j][r] + bj;
        int gm = mbase + r;
        if (!trans) {
          Y[(size_t)gm * H_ + o] = (bf16)val;
        } else {
          int bb = gm >> 12;             // / LK_
          int lk = gm & (LK_ - 1);
          Y[(size_t)bb * H_ * LK_ + (size_t)o * LK_ + lk] = (bf16)val;
        }
      }
    }
  }
}

// ============================================================
// Kernel 2: S = (q @ k^T) / inv_scale  (f32 logits into d_out)
// tiles: 4 x 128 (M) x 64 (N) = 32768 -> 4096 blocks.
// ============================================================
__global__ __launch_bounds__(256) void scores_kernel(
    const bf16* __restrict__ qb, const bf16* __restrict__ kb,
    const float* __restrict__ inv, float* __restrict__ S) {
  const int lane = threadIdx.x & 31;
  const int wave = threadIdx.x >> 5;
  int tile = blockIdx.x * 8 + wave;   // [0, 32768)
  int b = tile >> 13;
  int rem = tile & 8191;
  int tm = rem >> 6, tn = rem & 63;
  int m0 = tm * 32, n0 = tn * 64;

  const int lrow = lane & 15;
  const int aoff = (lane & 16) ? 16 : 0;
  const int boff = (lane & 16) ? 16 : 0;
  const bf16* aBase[MT];
  const bf16* bBase[NT];
#pragma unroll
  for (int i = 0; i < MT; ++i)
    aBase[i] = qb + (size_t)(b * LQ_ + m0 + 16 * i + lrow) * H_ + aoff;
#pragma unroll
  for (int j = 0; j < NT; ++j)
    bBase[j] = kb + (size_t)(b * LK_ + n0 + 16 * j + lrow) * H_ + boff;

  v8f acc[MT][NT] = {};
  for (int k = 0; k < H_; k += 32) {
    v16bf a[MT], bfr[NT];
#pragma unroll
    for (int i = 0; i < MT; ++i) a[i] = frag_a_bf16(aBase[i] + k);
#pragma unroll
    for (int j = 0; j < NT; ++j) bfr[j] = frag_b_bf16(bBase[j] + k);
#pragma unroll
    for (int i = 0; i < MT; ++i)
#pragma unroll
      for (int j = 0; j < NT; ++j) WMMA_BF16(acc[i][j], a[i], bfr[j]);
  }

  int n = lane & 15, hi = lane >> 4;
  size_t base = (size_t)b * LQ_ * LK_;
#pragma unroll
  for (int i = 0; i < MT; ++i)
#pragma unroll
    for (int j = 0; j < NT; ++j)
#pragma unroll
      for (int r = 0; r < 8; ++r) {
        size_t idx = base + (size_t)(m0 + 16 * i + 8 * hi + r) * LK_ +
                     (size_t)(n0 + 16 * j + n);
        S[idx] = acc[i][j][r] / inv[idx];
      }
}

// ============================================================
// Kernel 3: in-place row softmax over 4096-wide rows of S.
// ============================================================
__global__ __launch_bounds__(256) void softmax_kernel(float* __restrict__ S) {
  int row = blockIdx.x;                        // B_*LQ_ rows
  float* p = S + (size_t)row * LK_;
  int t = threadIdx.x;
  float v[16];
  float mx = -3.4e38f;
#pragma unroll
  for (int i = 0; i < 16; ++i) {
    v[i] = p[t + i * 256];
    mx = fmaxf(mx, v[i]);
  }
  __shared__ float red[256];
  red[t] = mx;
  __syncthreads();
  for (int s = 128; s > 0; s >>= 1) {
    if (t < s) red[t] = fmaxf(red[t], red[t + s]);
    __syncthreads();
  }
  mx = red[0];
  __syncthreads();
  float sum = 0.f;
#pragma unroll
  for (int i = 0; i < 16; ++i) {
    v[i] = __expf(v[i] - mx);
    sum += v[i];
  }
  red[t] = sum;
  __syncthreads();
  for (int s = 128; s > 0; s >>= 1) {
    if (t < s) red[t] += red[t + s];
    __syncthreads();
  }
  float rsum = 1.0f / red[0];
#pragma unroll
  for (int i = 0; i < 16; ++i) p[t + i * 256] = v[i] * rsum;
}

// ============================================================
// Kernel 4: O = P @ V via vT:[B][H][LK] bf16; P f32 -> bf16 on load.
// tiles: 4 x 128 (M) x 16 (N) = 8192 -> 1024 blocks.
// ============================================================
__global__ __launch_bounds__(256) void pv_kernel(
    const float* __restrict__ P, const bf16* __restrict__ vtb,
    float* __restrict__ O) {
  const int lane = threadIdx.x & 31;
  const int wave = threadIdx.x >> 5;
  int tile = blockIdx.x * 8 + wave;   // [0, 8192)
  int b = tile >> 11;
  int rem = tile & 2047;
  int tm = rem >> 4, tn = rem & 15;
  int m0 = tm * 32, n0 = tn * 64;

  const int lrow = lane & 15;
  const int aoff = (lane & 16) ? 16 : 0;
  const int boff = (lane & 16) ? 16 : 0;
  const float* aBase[MT];
  const bf16* bBase[NT];
#pragma unroll
  for (int i = 0; i < MT; ++i)
    aBase[i] = P + (size_t)b * LQ_ * LK_ +
               (size_t)(m0 + 16 * i + lrow) * LK_ + aoff;
#pragma unroll
  for (int j = 0; j < NT; ++j)
    bBase[j] = vtb + (size_t)b * H_ * LK_ +
               (size_t)(n0 + 16 * j + lrow) * LK_ + boff;

  v8f acc[MT][NT] = {};
  for (int k = 0; k < LK_; k += 32) {
    v16bf a[MT], bfr[NT];
#pragma unroll
    for (int i = 0; i < MT; ++i) a[i] = frag_a_f32(aBase[i] + k);
#pragma unroll
    for (int j = 0; j < NT; ++j) bfr[j] = frag_b_bf16(bBase[j] + k);
#pragma unroll
    for (int i = 0; i < MT; ++i)
#pragma unroll
      for (int j = 0; j < NT; ++j) WMMA_BF16(acc[i][j], a[i], bfr[j]);
  }

  int n = lane & 15, hi = lane >> 4;
#pragma unroll
  for (int i = 0; i < MT; ++i)
#pragma unroll
    for (int j = 0; j < NT; ++j)
#pragma unroll
      for (int r = 0; r < 8; ++r) {
        size_t om = (size_t)b * LQ_ + (size_t)(m0 + 16 * i + 8 * hi + r);
        O[om * H_ + (size_t)(n0 + 16 * j + n)] = acc[i][j][r];
      }
}

extern "C" void kernel_launch(void* const* d_in, const int* in_sizes, int n_in,
                              void* d_out, int out_size, void* d_ws,
                              size_t ws_size, hipStream_t stream) {
  const float* query = (const float*)d_in[0];
  const float* key_  = (const float*)d_in[1];
  const float* value = (const float*)d_in[2];
  const float* inv   = (const float*)d_in[3];
  const float* Wq = (const float*)d_in[4];
  const float* bq = (const float*)d_in[5];
  const float* Wk = (const float*)d_in[6];
  const float* bk = (const float*)d_in[7];
  const float* Wv = (const float*)d_in[8];
  const float* bv = (const float*)d_in[9];

  float* out = (float*)d_out;                       // [B,LQ,H]
  float* S   = out + (size_t)B_ * LQ_ * H_;         // [B,LQ,LK] softmax output

  bf16* qb  = (bf16*)d_ws;                          // [B*LQ, H] bf16
  bf16* kb  = qb + (size_t)B_ * LQ_ * H_;           // [B*LK, H] bf16
  bf16* vtb = kb + (size_t)B_ * LK_ * H_;           // [B, H, LK] bf16 (transposed)

  proj_kernel<<<1024, 256, 0, stream>>>(query, Wq, bq, qb, 0);
  proj_kernel<<<1024, 256, 0, stream>>>(key_, Wk, bk, kb, 0);
  proj_kernel<<<1024, 256, 0, stream>>>(value, Wv, bv, vtb, 1);
  scores_kernel<<<4096, 256, 0, stream>>>(qb, kb, inv, S);
  softmax_kernel<<<B_ * LQ_, 256, 0, stream>>>(S);
  pv_kernel<<<1024, 256, 0, stream>>>(S, vtb, out);
}
